// QuanvolutionHybrid_65481071403948
// MI455X (gfx1250) — compile-verified, matrix operand order checked by reading, not run on hardware
//
#include <hip/hip_runtime.h>
#include <hip/hip_bf16.h>

// ---------------------------------------------------------------------------
// Types for CDNA5 WMMA (wave32): D(16x16 f32) = A(16x32 bf16) x B(32x16 bf16) + C
// ---------------------------------------------------------------------------
typedef __attribute__((ext_vector_type(16))) __bf16 v16bf;
typedef __attribute__((ext_vector_type(8)))  __bf16 bf16x8;
typedef __attribute__((ext_vector_type(8)))  float  v8f;

__device__ __forceinline__ float bf2f(__bf16 b) {
    union { __bf16 b; unsigned short s; } i{b};
    union { unsigned u; float f; } o{(unsigned)i.s << 16};
    return o.f;
}

__device__ __forceinline__ __bf16 f2bf(float f) {
    union { float f; unsigned u; } i{f};
    unsigned u = i.u;
    unsigned r = (u + 0x7FFFu + ((u >> 16) & 1u)) >> 16;   // round-to-nearest-even
    union { unsigned short s; __bf16 b; } o{(unsigned short)r};
    return o.b;
}

// Per-lane fragment load matching the ISA 16-bit A/B layout:
// lanes 0-15  : K = kb+[0..7]  and kb+[16..23]   (M or N = lane)
// lanes 16-31 : K = kb+[8..15] and kb+[24..31]   (M or N = lane-16)
__device__ __forceinline__ v16bf load_frag_g(const __bf16* __restrict__ row, int kb, int half) {
    union { v16bf v; bf16x8 h[2]; } f;
    f.h[0] = *reinterpret_cast<const bf16x8*>(row + kb + half * 8);
    f.h[1] = *reinterpret_cast<const bf16x8*>(row + kb + 16 + half * 8);
    return f.v;
}

// Same fragment pattern out of an LDS row of 32 bf16 (64 B stride)
__device__ __forceinline__ v16bf load_frag_lds(const __bf16* row, int half) {
    union { v16bf v; bf16x8 h[2]; } f;
    f.h[0] = *reinterpret_cast<const bf16x8*>(row + half * 8);
    f.h[1] = *reinterpret_cast<const bf16x8*>(row + 16 + half * 8);
    return f.v;
}

// gfx1250 async global -> LDS copy, 16 bytes per lane (tracked by ASYNCcnt)
__device__ __forceinline__ void async_b128_to_lds(unsigned lds_off, const void* gaddr) {
    asm volatile("global_load_async_to_lds_b128 %0, %1, off"
                 :: "v"(lds_off), "v"(gaddr) : "memory");
}

// ---------------------------------------------------------------------------
// Quanvolution: 2x2 stride-2 conv 1->4ch, flatten NCHW to (B,784), store bf16
// padded to K=800 (multiple of 32) for the WMMA K-loop.
// ---------------------------------------------------------------------------
__global__ __launch_bounds__(256) void conv_feat_kernel(
    const float* __restrict__ x, const float* __restrict__ cw,
    const float* __restrict__ cb, __bf16* __restrict__ featB, int B)
{
    int idx = blockIdx.x * 256 + threadIdx.x;
    if (idx >= B * 800) return;
    int b = idx / 800, k = idx % 800;
    if (k < 784) {
        int c = k / 196, rem = k % 196, i = rem / 14, j = rem % 14;
        const float* xp = x + (size_t)b * 784 + (2 * i) * 28 + 2 * j;
        float v = xp[0] * cw[c * 4 + 0] + xp[1]  * cw[c * 4 + 1] +
                  xp[28] * cw[c * 4 + 2] + xp[29] * cw[c * 4 + 3] + cb[c];
        featB[idx] = f2bf(v);
    } else {
        featB[idx] = f2bf(0.f);
    }
}

// f32 -> bf16 with optional zero padding to row stride ldd
__global__ __launch_bounds__(256) void cvt_pad_kernel(
    const float* __restrict__ src, __bf16* __restrict__ dst,
    int rows, int K, int ldd)
{
    int idx = blockIdx.x * 256 + threadIdx.x;
    if (idx >= rows * ldd) return;
    int r = idx / ldd, k = idx % ldd;
    dst[idx] = (k < K) ? f2bf(src[(size_t)r * K + k]) : f2bf(0.f);
}

// Row squared-norms of a bf16 matrix (padding is zero so it is harmless)
__global__ __launch_bounds__(256) void rownorm2_kernel(
    const __bf16* __restrict__ M, int ld, float* __restrict__ out)
{
    int row = blockIdx.x, tid = threadIdx.x;
    __shared__ float red[256];
    float s = 0.f;
    const __bf16* r = M + (size_t)row * ld;
    for (int k = tid; k < ld; k += 256) { float v = bf2f(r[k]); s += v * v; }
    red[tid] = s; __syncthreads();
    for (int off = 128; off > 0; off >>= 1) {
        if (tid < off) red[tid] += red[tid + off];
        __syncthreads();
    }
    if (tid == 0) out[row] = red[0];
}

// ---------------------------------------------------------------------------
// WMMA GEMM, block tile 256x64, wave tile 32x64.
//   - B tile (64 rows x 32 K = 4KB) staged once per block per K-step via
//     global_load_async_to_lds_b128 (double buffered, ASYNCcnt + barrier),
//     shared by all 8 waves.
//   - A fragments loaded directly from global (distinct rows per wave).
//   - 8 WMMAs per K-step per wave (2 A frags x 4 B frags).
// MODE 0: out = relu(acc + bias[n])        (hidden layers)
// MODE 1: out = exp(2*acc - f2[m] - p2[n]) (RBF kernel layer, GAMMA = 1)
// ---------------------------------------------------------------------------
template <int MODE>
__global__ __launch_bounds__(256) void gemm_wmma_kernel(
    const __bf16* __restrict__ A,  int lda,
    const __bf16* __restrict__ Bm, int ldb,
    const float* __restrict__ bias,
    const float* __restrict__ f2v,
    const float* __restrict__ p2v,
    __bf16* __restrict__ C, int M, int N, int K)
{
    __shared__ __bf16 sB[2][64 * 32];            // 2 x 4KB double buffer

    const int tid   = threadIdx.x;
    const int lane  = tid & 31;
    const int wv    = tid >> 5;                  // wave in block: 0..7
    const int mblks = M >> 8;                    // 256 rows per block
    const int mblk  = blockIdx.x % mblks;
    const int nstrip = blockIdx.x / mblks;
    const int half  = lane >> 4;
    const int lrow  = lane & 15;

    // A rows for this wave (32 rows)
    const __bf16* rowA0 = A + (size_t)(mblk * 256 + wv * 32 + lrow) * lda;
    const __bf16* rowA1 = A + (size_t)(mblk * 256 + wv * 32 + 16 + lrow) * lda;

    // B staging: thread t copies 16B: row = t/4 (0..63), segment = t%4
    const int srow = tid >> 2, sseg = tid & 3;
    const __bf16* gB = Bm + (size_t)(nstrip * 64 + srow) * ldb + sseg * 8;
    const unsigned ldsDst0 = (unsigned)(size_t)&sB[0][srow * 32 + sseg * 8];
    const unsigned ldsDst1 = (unsigned)(size_t)&sB[1][srow * 32 + sseg * 8];

    v8f acc[2][4] = {};

    // Prologue: stage kb = 0 into buffer 0
    async_b128_to_lds(ldsDst0, gB);

    int buf = 0;
    for (int kb = 0; kb < K; kb += 32) {
        if (kb + 32 < K) {
            async_b128_to_lds(buf ? ldsDst0 : ldsDst1, gB + kb + 32);
            asm volatile("s_wait_asynccnt 0x1" ::: "memory");
        } else {
            asm volatile("s_wait_asynccnt 0x0" ::: "memory");
        }
        __syncthreads();                          // current tile visible to all

        v16bf a0 = load_frag_g(rowA0, kb, half);
        v16bf a1 = load_frag_g(rowA1, kb, half);
        __builtin_prefetch(rowA0 + kb + 32, 0, 1);
        __builtin_prefetch(rowA1 + kb + 32, 0, 1);

        const __bf16* sb = sB[buf];
#pragma unroll
        for (int j = 0; j < 4; ++j) {
            v16bf b = load_frag_lds(sb + (j * 16 + lrow) * 32, half);
            acc[0][j] = __builtin_amdgcn_wmma_f32_16x16x32_bf16(
                false, a0, false, b, (short)0, acc[0][j], false, false);
            acc[1][j] = __builtin_amdgcn_wmma_f32_16x16x32_bf16(
                false, a1, false, b, (short)0, acc[1][j], false, false);
        }
        __syncthreads();                          // safe to overwrite this buffer
        buf ^= 1;
    }

    // C/D layout: lanes 0-15 hold M = 0..7 (VGPR r), lanes 16-31 hold M = 8..15
    const int nbase = nstrip * 64 + lrow;
#pragma unroll
    for (int i = 0; i < 2; ++i) {
        const int mrow0 = mblk * 256 + wv * 32 + i * 16 + half * 8;
#pragma unroll
        for (int j = 0; j < 4; ++j) {
            const int col = nbase + j * 16;
            const float cterm = (MODE == 0) ? bias[col] : p2v[col];
#pragma unroll
            for (int r = 0; r < 8; ++r) {
                const int row = mrow0 + r;
                float v = acc[i][j][r];
                if (MODE == 0) {
                    v = fmaxf(v + cterm, 0.f);
                } else {
                    v = __expf(2.0f * v - f2v[row] - cterm);  // exp(-||f-p||^2)
                }
                C[(size_t)row * N + col] = f2bf(v);
            }
        }
    }
}

// ---------------------------------------------------------------------------
// Head: logits = H(bf16) @ Wout^T + bout, then log_softmax over 10 classes.
// ---------------------------------------------------------------------------
__global__ __launch_bounds__(256) void head_kernel(
    const __bf16* __restrict__ H, const float* __restrict__ Wout,
    const float* __restrict__ bout, float* __restrict__ out, int K)
{
    const int b = blockIdx.x, tid = threadIdx.x;
    float p[10];
#pragma unroll
    for (int c = 0; c < 10; ++c) p[c] = 0.f;
    const __bf16* hrow = H + (size_t)b * K;
    for (int k = tid; k < K; k += 256) {
        float h = bf2f(hrow[k]);
#pragma unroll
        for (int c = 0; c < 10; ++c) p[c] += h * Wout[(size_t)c * K + k];
    }
    __shared__ float red[256];
    __shared__ float logits[10];
    for (int c = 0; c < 10; ++c) {
        red[tid] = p[c]; __syncthreads();
        for (int off = 128; off > 0; off >>= 1) {
            if (tid < off) red[tid] += red[tid + off];
            __syncthreads();
        }
        if (tid == 0) logits[c] = red[0] + bout[c];
        __syncthreads();
    }
    if (tid == 0) {
        float mx = logits[0];
        for (int c = 1; c < 10; ++c) mx = fmaxf(mx, logits[c]);
        float s = 0.f;
        for (int c = 0; c < 10; ++c) s += __expf(logits[c] - mx);
        float lse = mx + __logf(s);
        for (int c = 0; c < 10; ++c) out[(size_t)b * 10 + c] = logits[c] - lse;
    }
}

// ---------------------------------------------------------------------------
// Launcher
// ---------------------------------------------------------------------------
extern "C" void kernel_launch(void* const* d_in, const int* in_sizes, int n_in,
                              void* d_out, int out_size, void* d_ws, size_t ws_size,
                              hipStream_t stream)
{
    (void)in_sizes; (void)n_in; (void)out_size; (void)ws_size;

    const float* x      = (const float*)d_in[0];
    const float* conv_w = (const float*)d_in[1];
    const float* conv_b = (const float*)d_in[2];
    const float* protos = (const float*)d_in[3];
    const float* W0 = (const float*)d_in[4];
    const float* b0 = (const float*)d_in[5];
    const float* W1 = (const float*)d_in[6];
    const float* b1 = (const float*)d_in[7];
    const float* W2 = (const float*)d_in[8];
    const float* b2 = (const float*)d_in[9];
    const float* Wout = (const float*)d_in[10];
    const float* bout = (const float*)d_in[11];
    float* out = (float*)d_out;

    constexpr int Bsz = 8192, P = 2048, F = 784, FP = 800;  // FP: 784 padded to /32

    char* w = (char*)d_ws;
    auto alloc = [&](size_t bytes) -> void* {
        void* p = (void*)w;
        w += (bytes + 255) & ~size_t(255);
        return p;
    };
    __bf16* featB  = (__bf16*)alloc((size_t)Bsz * FP * 2);  // 12.8 MB
    __bf16* protoB = (__bf16*)alloc((size_t)P   * FP * 2);  //  3.2 MB
    __bf16* W0b    = (__bf16*)alloc((size_t)P * P * 2);     //  8.0 MB
    __bf16* W1b    = (__bf16*)alloc((size_t)P * P * 2);
    __bf16* W2b    = (__bf16*)alloc((size_t)P * P * 2);
    float*  f2v    = (float*)alloc((size_t)Bsz * 4);
    float*  p2v    = (float*)alloc((size_t)P * 4);
    __bf16* H0     = (__bf16*)alloc((size_t)Bsz * P * 2);   // 32 MB ping
    __bf16* H1     = (__bf16*)alloc((size_t)Bsz * P * 2);   // 32 MB pong

    // Stage bf16 operands
    conv_feat_kernel<<<(Bsz * FP + 255) / 256, 256, 0, stream>>>(x, conv_w, conv_b, featB, Bsz);
    cvt_pad_kernel<<<(P * FP + 255) / 256, 256, 0, stream>>>(protos, protoB, P, F, FP);
    cvt_pad_kernel<<<(P * P) / 256, 256, 0, stream>>>(W0, W0b, P, P, P);
    cvt_pad_kernel<<<(P * P) / 256, 256, 0, stream>>>(W1, W1b, P, P, P);
    cvt_pad_kernel<<<(P * P) / 256, 256, 0, stream>>>(W2, W2b, P, P, P);
    rownorm2_kernel<<<Bsz, 256, 0, stream>>>(featB, FP, f2v);
    rownorm2_kernel<<<P, 256, 0, stream>>>(protoB, FP, p2v);

    // GEMM grid: block tile 256x64 -> (8192/256)*(2048/64) = 1024 blocks
    const int gemm_blocks = (Bsz / 256) * (P / 64);

    // RBF kernel layer: ker = exp(2*feat.proto - ||f||^2 - ||p||^2)
    gemm_wmma_kernel<1><<<gemm_blocks, 256, 0, stream>>>(
        featB, FP, protoB, FP, nullptr, f2v, p2v, H0, Bsz, P, FP);

    // Three hidden layers: h = relu(h @ W^T + b)
    gemm_wmma_kernel<0><<<gemm_blocks, 256, 0, stream>>>(
        H0, P, W0b, P, b0, nullptr, nullptr, H1, Bsz, P, P);
    gemm_wmma_kernel<0><<<gemm_blocks, 256, 0, stream>>>(
        H1, P, W1b, P, b1, nullptr, nullptr, H0, Bsz, P, P);
    gemm_wmma_kernel<0><<<gemm_blocks, 256, 0, stream>>>(
        H0, P, W2b, P, b2, nullptr, nullptr, H1, Bsz, P, P);

    // Output head + log_softmax
    head_kernel<<<Bsz, 256, 0, stream>>>(H1, Wout, bout, out, P);
}
